// window_attention_44375602102362
// MI455X (gfx1250) — compile-verified, matrix operand order checked by reading, not run on hardware
//
#include <hip/hip_runtime.h>
#include <hip/hip_bf16.h>

typedef __attribute__((ext_vector_type(16))) _Float16 v16h;
typedef __attribute__((ext_vector_type(8)))  _Float16 v8h;
typedef __attribute__((ext_vector_type(8)))  float    v8f;

#define NHEADS 4
#define NN 49
#define DD 128
#define NPAD 64
#define ATT_SCALE 0.17677669529663687f   // 1/sqrt(32)

// Load one 16x32 (f16) WMMA operand fragment from row-major storage.
// For A: row0+m = matrix row, K along the row. For B: pass transposed
// storage (row = output column n), identical lane layout.
// CDNA5 16-bit layout: lane l (m=l&15, h=l>>4): elements 0..7 = K[8h..8h+7],
// elements 8..15 = K[16+8h..16+8h+7]  -> two contiguous 16B chunks.
__device__ __forceinline__ v16h load_frag(const _Float16* __restrict__ base,
                                          int row0, int pitch, int kbase, int lane) {
  int m  = lane & 15;
  int hh = lane >> 4;
  const _Float16* p = base + (size_t)(row0 + m) * pitch + kbase + hh * 8;
  union { v16h v; v8h h[2]; } u;
  u.h[0] = *(const v8h*)(p);
  u.h[1] = *(const v8h*)(p + 16);
  return u.v;
}

__device__ __forceinline__ v8f wmma_f16(v16h a, v16h b, v8f c) {
  return __builtin_amdgcn_wmma_f32_16x16x32_f16(false, a, false, b, (short)0, c,
                                                false, false);
}

// One-time weight transpose + f32->f16 convert into workspace.
__global__ void prep_weights_kernel(const float* __restrict__ wqkv,
                                    const float* __restrict__ wproj,
                                    _Float16* __restrict__ wqkvT,
                                    _Float16* __restrict__ wprojT) {
  int i = blockIdx.x * blockDim.x + threadIdx.x;
  if (i < 384 * 128) {          // wqkvT[n][k] = wqkv[k][n],  n<384, k<128
    int n = i >> 7, k = i & 127;
    wqkvT[i] = (_Float16)wqkv[k * 384 + n];
  }
  if (i < 128 * 128) {          // wprojT[n][k] = wproj[k][n]
    int n = i >> 7, k = i & 127;
    wprojT[i] = (_Float16)wproj[k * 128 + n];
  }
}

__global__ __launch_bounds__(256, 1) void win_attn_kernel(
    const float* __restrict__ x, const float* __restrict__ mask,
    const _Float16* __restrict__ wqkvT, const _Float16* __restrict__ wprojT,
    const float* __restrict__ bproj, float* __restrict__ out) {
  __shared__ _Float16 xsO[NPAD * DD];              // x(f16) in stage A; O in B2/C
  __shared__ _Float16 qk[2][NHEADS * 64 * 32];     // q,k row-major [H][64][32]
  __shared__ _Float16 vT[NHEADS * 32 * 64];        // v transposed [H][32][64]
  __shared__ float    SP[NHEADS * 64 * 64];        // S f32; P f16 overlaid in place
  __shared__ float    mask_s[NPAD * NPAD];         // zero-padded 64x64 window mask

  const int tid  = threadIdx.x;
  const int lane = tid & 31;
  // Wave index as a proven-uniform scalar: turns tiling branches/addresses
  // into SALU + s_cbranch instead of per-lane EXEC juggling.
  const int wvu  = __builtin_amdgcn_readfirstlane(tid >> 5);   // 0..7
  const int b    = blockIdx.x;        // one window per block
  const int m0   = (lane >> 4) * 8;   // C-frag row base for this half-wave
  const int nn   = lane & 15;         // C-frag column

  // ---- load x[b] as f16 into LDS (rows >=49 zeroed) ----
  const float4* xv = (const float4*)(x + (size_t)b * (NN * DD));
  for (int i = tid; i < NPAD * 32; i += 256) {
    int row = i >> 5, q4 = i & 31;
    _Float16* d = &xsO[row * DD + q4 * 4];
    if (row < NN) {
      float4 t = xv[row * 32 + q4];
      d[0] = (_Float16)t.x; d[1] = (_Float16)t.y;
      d[2] = (_Float16)t.z; d[3] = (_Float16)t.w;
    } else {
      d[0] = d[1] = d[2] = d[3] = (_Float16)0.f;
    }
  }
  // ---- window mask, zero-padded to 64x64 (window = b % 64) ----
  const float* mp = mask + (size_t)(b & 63) * (NN * NN);
  for (int i = tid; i < NPAD * NPAD; i += 256) {
    int row = i >> 6, col = i & 63;
    mask_s[i] = (row < NN && col < NN) ? mp[row * NN + col] : 0.f;
  }
  __syncthreads();

  // ---- Stage A: qkv = x @ w_qkv   (M=64, K=128, N=384) ----
#pragma unroll
  for (int ni = 0; ni < 3; ++ni) {
    const int nt = wvu + ni * 8;                  // scalar
    v16h bf[4];
#pragma unroll
    for (int ks = 0; ks < 4; ++ks)
      bf[ks] = load_frag(wqkvT, nt * 16, DD, ks * 32, lane);
    const int cb    = nt * 16;                    // scalar
    const int seg   = cb >> 7;                    // 0=q 1=k 2=v (scalar)
    const int hh    = (cb & 127) >> 5;            // head (scalar)
    const int dbase = cb & 31;                    // 0 or 16 (scalar)
#pragma unroll
    for (int mt = 0; mt < 4; ++mt) {
      v8f c = {};
#pragma unroll
      for (int ks = 0; ks < 4; ++ks)
        c = wmma_f16(load_frag(xsO, mt * 16, DD, ks * 32, lane), bf[ks], c);
      if (seg < 2) {                               // scalar branch
        _Float16* dst = &qk[seg][(hh * 64 + mt * 16 + m0) * 32 + dbase + nn];
#pragma unroll
        for (int r = 0; r < 8; ++r) dst[r * 32] = (_Float16)c[r];
      } else {
        union { v8h v; _Float16 e[8]; } pk;
#pragma unroll
        for (int r = 0; r < 8; ++r) pk.e[r] = (_Float16)c[r];
        // 8 contiguous halves -> single 16B LDS store
        *(v8h*)&vT[(hh * 32 + dbase + nn) * 64 + mt * 16 + m0] = pk.v;
      }
    }
  }
  __syncthreads();

  // ---- Stage B1: S = (q @ k^T) * scale + mask  (per head, K=32 = 1 WMMA) ----
  {
    const int hh = wvu >> 1;                      // scalar head
    const int p  = wvu & 1;
    const _Float16* qb = &qk[0][hh * 64 * 32];
    const _Float16* kb = &qk[1][hh * 64 * 32];
    float* Sh = &SP[hh * 64 * 64];
#pragma unroll
    for (int mi = 0; mi < 2; ++mi) {
      int mt = p * 2 + mi;
      v16h af = load_frag(qb, mt * 16, 32, 0, lane);
#pragma unroll
      for (int nt = 0; nt < 4; ++nt) {
        v8f c = {};
        c = wmma_f16(af, load_frag(kb, nt * 16, 32, 0, lane), c);
        const int col = nt * 16 + nn;
#pragma unroll
        for (int r = 0; r < 8; ++r) {
          int row = mt * 16 + m0 + r;
          Sh[row * 64 + col] = c[r] * ATT_SCALE + mask_s[row * 64 + col];
        }
      }
    }
  }
  __syncthreads();

  // ---- Softmax: one thread per (head,row); write P f16 in place over S f32.
  // Forward order is safe: f16 write at byte 2c never passes the f32 read at 4c.
  {
    const int hh = tid >> 6, row = tid & 63;
    float* srow = &SP[hh * 4096 + row * 64];
    float mx = -1e30f;
    for (int c = 0; c < NN; ++c) mx = fmaxf(mx, srow[c]);
    float sum = 0.f;
    for (int c = 0; c < NN; ++c) sum += __expf(srow[c] - mx);
    float rinv = 1.0f / sum;
    _Float16* prow = (_Float16*)srow;
    for (int c = 0; c < 64; ++c) {
      float v = (c < NN) ? __expf(srow[c] - mx) * rinv : 0.f;
      asm volatile("" ::: "memory");   // keep read-before-write ordering
      prow[c] = (_Float16)v;
    }
  }
  __syncthreads();

  // ---- Stage B2: O = P @ v  (per head: M=64, K=64, N=32) ----
  {
    const int hh = wvu >> 1;
    const int p  = wvu & 1;
    const _Float16* Pb = (const _Float16*)&SP[hh * 4096];  // row pitch 128 halves
    const _Float16* vb = &vT[hh * 32 * 64];
#pragma unroll
    for (int mi = 0; mi < 2; ++mi) {
      int mt = p * 2 + mi;
#pragma unroll
      for (int nt = 0; nt < 2; ++nt) {
        v8f c = {};
#pragma unroll
        for (int ks = 0; ks < 2; ++ks)
          c = wmma_f16(load_frag(Pb, mt * 16, 128, ks * 32, lane),
                       load_frag(vb, nt * 16, 64,  ks * 32, lane), c);
        _Float16* dst = &xsO[(mt * 16 + m0) * DD + hh * 32 + nt * 16 + nn];
#pragma unroll
        for (int r = 0; r < 8; ++r) dst[r * DD] = (_Float16)c[r];  // O reuses xs
      }
    }
  }
  __syncthreads();

  // ---- Stage C: out = O @ w_proj + b_proj  (M=64, K=128, N=128) ----
  {
    const int nt = wvu;                      // 8 waves x 16 cols = 128
    const float bb = bproj[nt * 16 + nn];
    v16h bf[4];
#pragma unroll
    for (int ks = 0; ks < 4; ++ks)
      bf[ks] = load_frag(wprojT, nt * 16, DD, ks * 32, lane);
    float* ob = out + (size_t)b * (NN * DD);
    const int col = nt * 16 + nn;
#pragma unroll
    for (int mt = 0; mt < 4; ++mt) {
      v8f c = {};
#pragma unroll
      for (int ks = 0; ks < 4; ++ks)
        c = wmma_f16(load_frag(xsO, mt * 16, DD, ks * 32, lane), bf[ks], c);
#pragma unroll
      for (int r = 0; r < 8; ++r) {
        int row = mt * 16 + m0 + r;
        if (row < NN) ob[row * DD + col] = c[r] + bb;
      }
    }
  }
}

extern "C" void kernel_launch(void* const* d_in, const int* in_sizes, int n_in,
                              void* d_out, int out_size, void* d_ws, size_t ws_size,
                              hipStream_t stream) {
  const float* x     = (const float*)d_in[0];   // [8192,49,128]
  const float* mask  = (const float*)d_in[1];   // [64,49,49]
  const float* wqkv  = (const float*)d_in[2];   // [128,384]
  const float* wproj = (const float*)d_in[3];   // [128,128]
  const float* bproj = (const float*)d_in[4];   // [128]

  _Float16* wqkvT  = (_Float16*)d_ws;                                   // 98304 B
  _Float16* wprojT = (_Float16*)((char*)d_ws + 384 * 128 * sizeof(_Float16));

  prep_weights_kernel<<<192, 256, 0, stream>>>(wqkv, wproj, wqkvT, wprojT);
  win_attn_kernel<<<8192, 256, 0, stream>>>(x, mask, wqkvT, wprojT, bproj,
                                            (float*)d_out);
}
